// FastR2D2DPClassification_76811195122066
// MI455X (gfx1250) — matrix operand order repaired and checked by reading, not compiled, not run on hardware
//
#include <hip/hip_runtime.h>
#include <stdint.h>

// ---------------- problem constants ----------------
#define EMB   768
#define HID   3072
#define TOT   66
#define SEQL  512
#define NODES 1023
#define NBATCH 128
#define ROWS  (NBATCH * NODES)   // 130944

// ---------------- tiling ----------------
#define MT 128         // rows per block
#define NT 128         // hid cols per chunk
#define NCHUNK (HID / NT)   // 24
#define KBLK 12             // 64-wide k-slices for GEMM1 (768/64)
#define K2 (NT / 32)        // 4 k-steps for GEMM2

// LDS layout (ushort strides / byte offsets); dynamic LDS, 305152 B total
#define E_STRIDE  776        // 768 + 8 pad
#define W1_STRIDE 72         // 64 + 8 pad
#define H_STRIDE  136        // 128 + 8 pad
#define W2_STRIDE 136

#define E_OFF_B   0u         // 128*776*2         = 198656
#define W1_OFF_B  198656u    // 2 bufs * 128*72*2 = 36864
#define W1_BUF_B  18432u
#define H_OFF_B   235520u    // 128*136*2         = 34816
#define W2_OFF_B  270336u    // 128*136*2         = 34816
#define LDS_BYTES 305152u

typedef __attribute__((ext_vector_type(16))) __bf16 v16bf;
typedef __attribute__((ext_vector_type(8)))  float  v8f;

union Frag { uint4 u[2]; v16bf v; };

__device__ __forceinline__ unsigned short f2bf(float f) {
  uint32_t u = __float_as_uint(f);
  u += 0x7FFFu + ((u >> 16) & 1u);         // round-to-nearest-even
  return (unsigned short)(u >> 16);
}

// async 16-byte global -> LDS copy (CDNA5 async path, tracked by ASYNCcnt)
__device__ __forceinline__ void async_cp16(uint32_t lds_addr, const void* g) {
  asm volatile("global_load_async_to_lds_b128 %0, %1, off"
               :: "v"(lds_addr), "v"(g) : "memory");
}
__device__ __forceinline__ void wait_async0() {
  asm volatile("s_wait_asynccnt 0" ::: "memory");
}

// ---------------- prep: fp32 -> bf16 conversions ----------------
__global__ void cvt_emb_kernel(const float* __restrict__ in,
                               unsigned short* __restrict__ out, long long n) {
  long long i = (long long)blockIdx.x * blockDim.x + threadIdx.x;
  long long stride = (long long)gridDim.x * blockDim.x;
  for (; i < n; i += stride) out[i] = f2bf(in[i]);
}

__global__ void cvt_w1t_kernel(const float* __restrict__ W1,
                               unsigned short* __restrict__ W1T) {
  int idx = blockIdx.x * 256 + threadIdx.x;       // 768*3072 total
  if (idx < EMB * HID) {
    int k = idx / HID, n = idx % HID;             // W1 is [k][n]
    W1T[(size_t)n * EMB + k] = f2bf(W1[idx]);     // W1T is [n][k], k contiguous
  }
}

__global__ void cvt_w2t_kernel(const float* __restrict__ W2,
                               unsigned short* __restrict__ W2T) {
  int idx = blockIdx.x * 256 + threadIdx.x;       // 128*3072 total
  if (idx < 128 * HID) {
    int n = idx / HID, k = idx % HID;             // W2 is [k][n], n<66
    float v = (n < TOT) ? W2[(size_t)k * TOT + n] : 0.0f;
    W2T[idx] = f2bf(v);                           // W2T is [n][k] padded to n=128
  }
}

// ---------------- fused MLP: gelu(E@W1+b1)@W2+b2 -> sigmoid -> probs ----------------
__global__ __launch_bounds__(256) void fused_mlp_kernel(
    const unsigned short* __restrict__ Ebf,   // [ROWS][768]   bf16
    const unsigned short* __restrict__ W1T,   // [3072][768]   bf16, transposed
    const unsigned short* __restrict__ W2T,   // [128][3072]   bf16, transposed+padded
    const float* __restrict__ b1,
    const float* __restrict__ b2,
    float* __restrict__ probs)                // [ROWS][66]    f32
{
  extern __shared__ char smem[];
  unsigned short* lds = (unsigned short*)smem;
  const uint32_t lds_base = (uint32_t)(uintptr_t)smem;

  const int tid  = threadIdx.x;
  const int lane = tid & 31;
  const int w    = tid >> 5;
  const int rw   = w >> 1;                  // row quarter 0..3 (32 rows each)
  const int cw   = w & 1;                   // col half    0..1 (64 cols each)
  const int row0 = blockIdx.x * MT;

  const uint32_t selA = (lane & 16) ? 8u : 0u;   // A-frag K half select
  const uint32_t klo  = (lane & 16) ? 8u : 0u;   // B-frag K half select
  const int colL = lane & 15;
  const int rA0  = 32 * rw + colL;               // A row, stripe 0
  const int rA1  = rA0 + 16;                     // A row, stripe 1

  // ---- async stage E tile: 128 rows x 768 bf16 = 12288 x 16B chunks ----
  {
    const unsigned short* gE = Ebf + (size_t)row0 * EMB;
#pragma unroll
    for (int i = 0; i < 48; ++i) {
      int cc  = tid + i * 256;
      int row = cc / 96;
      int col = (cc % 96) * 8;
      async_cp16(lds_base + E_OFF_B + (uint32_t)(row * E_STRIDE + col) * 2u,
                 gE + (size_t)row * EMB + col);
    }
  }

  const v8f vzero = {0.f, 0.f, 0.f, 0.f, 0.f, 0.f, 0.f, 0.f};
  v8f acc2[2][4];
#pragma unroll
  for (int s = 0; s < 2; ++s)
#pragma unroll
    for (int t = 0; t < 4; ++t) acc2[s][t] = vzero;

  for (int ch = 0; ch < NCHUNK; ++ch) {
    const int n0 = ch * NT;

    // ---- async stage W2T chunk: 128 x 128 bf16 = 2048 x 16B chunks ----
#pragma unroll
    for (int i = 0; i < 8; ++i) {
      int cc = tid + i * 256;
      int n  = cc >> 4;
      int kk = (cc & 15) * 8;
      async_cp16(lds_base + W2_OFF_B + (uint32_t)(n * W2_STRIDE + kk) * 2u,
                 W2T + (size_t)n * HID + n0 + kk);
    }
    // ---- async stage W1T k-slice 0 (64 wide) -> buf 0: 1024 chunks ----
#pragma unroll
    for (int i = 0; i < 4; ++i) {
      int cc = tid + i * 256;
      int n  = cc >> 3;
      int kk = (cc & 7) * 8;
      async_cp16(lds_base + W1_OFF_B + (uint32_t)(n * W1_STRIDE + kk) * 2u,
                 W1T + (size_t)(n0 + n) * EMB + kk);
    }
    wait_async0();
    __syncthreads();

    v8f acc1[2][4];
#pragma unroll
    for (int s = 0; s < 2; ++s)
#pragma unroll
      for (int t = 0; t < 4; ++t) acc1[s][t] = vzero;

    // ---- GEMM1: 128x128 tile over K=768, double-buffered 64-wide W1 slices ----
    for (int kb = 0; kb < KBLK; ++kb) {
      const uint32_t buf = (uint32_t)(kb & 1);
      if (kb + 1 < KBLK) {             // prefetch next 64-wide slice
        const uint32_t nbuf = buf ^ 1u;
#pragma unroll
        for (int i = 0; i < 4; ++i) {
          int cc = tid + i * 256;
          int n  = cc >> 3;
          int kk = (cc & 7) * 8;
          async_cp16(lds_base + W1_OFF_B + nbuf * W1_BUF_B +
                         (uint32_t)(n * W1_STRIDE + kk) * 2u,
                     W1T + (size_t)(n0 + n) * EMB + (kb + 1) * 64 + kk);
        }
      }
      const unsigned short* w1b = lds + (W1_OFF_B / 2) + buf * (W1_BUF_B / 2);
#pragma unroll
      for (int j = 0; j < 2; ++j) {    // two 32-wide k-steps per slice
        const int kg = kb * 64 + j * 32;
        Frag a0, a1;
        const unsigned short* ap0 = lds + rA0 * E_STRIDE + kg + selA;
        const unsigned short* ap1 = lds + rA1 * E_STRIDE + kg + selA;
        a0.u[0] = *(const uint4*)ap0;
        a0.u[1] = *(const uint4*)(ap0 + 16);
        a1.u[0] = *(const uint4*)ap1;
        a1.u[1] = *(const uint4*)(ap1 + 16);
#pragma unroll
        for (int t = 0; t < 4; ++t) {
          Frag b;
          const unsigned short* bp =
              w1b + (uint32_t)((cw * 64 + t * 16 + colL) * W1_STRIDE) + j * 32 + klo;
          b.u[0] = *(const uint4*)bp;
          b.u[1] = *(const uint4*)(bp + 16);
          acc1[0][t] = __builtin_amdgcn_wmma_f32_16x16x32_bf16(
              false, a0.v, false, b.v, (short)0, acc1[0][t], false, false);
          acc1[1][t] = __builtin_amdgcn_wmma_f32_16x16x32_bf16(
              false, a1.v, false, b.v, (short)0, acc1[1][t], false, false);
        }
      }
      if (kb + 1 < KBLK) wait_async0();
      __syncthreads();
    }

    // ---- bias + gelu(tanh) -> bf16 H tile in LDS ----
#pragma unroll
    for (int s = 0; s < 2; ++s) {
#pragma unroll
      for (int t = 0; t < 4; ++t) {
        const int col = cw * 64 + t * 16 + colL;
        const float bias = b1[n0 + col];
#pragma unroll
        for (int i = 0; i < 8; ++i) {
          float x  = acc1[s][t][i] + bias;
          float x3 = x * x * x;
          float g  = 0.5f * x * (1.0f + tanhf(0.7978845608028654f * (x + 0.044715f * x3)));
          int rr = 32 * rw + 16 * s + i + ((lane >> 4) & 1) * 8;
          lds[H_OFF_B / 2 + (uint32_t)(rr * H_STRIDE + col)] = f2bf(g);
        }
      }
    }
    __syncthreads();

    // ---- GEMM2: accumulate logits (128 x 128padded) over this 128-wide K chunk ----
#pragma unroll
    for (int ks2 = 0; ks2 < K2; ++ks2) {
      Frag a0, a1;
      const unsigned short* ap0 = lds + H_OFF_B / 2 + rA0 * H_STRIDE + ks2 * 32 + selA;
      const unsigned short* ap1 = lds + H_OFF_B / 2 + rA1 * H_STRIDE + ks2 * 32 + selA;
      a0.u[0] = *(const uint4*)ap0;
      a0.u[1] = *(const uint4*)(ap0 + 16);
      a1.u[0] = *(const uint4*)ap1;
      a1.u[1] = *(const uint4*)(ap1 + 16);
#pragma unroll
      for (int t = 0; t < 4; ++t) {
        Frag b;
        const unsigned short* bp = lds + W2_OFF_B / 2 +
                                   (uint32_t)((cw * 64 + t * 16 + colL) * W2_STRIDE) +
                                   ks2 * 32 + klo;
        b.u[0] = *(const uint4*)bp;
        b.u[1] = *(const uint4*)(bp + 16);
        acc2[0][t] = __builtin_amdgcn_wmma_f32_16x16x32_bf16(
            false, a0.v, false, b.v, (short)0, acc2[0][t], false, false);
        acc2[1][t] = __builtin_amdgcn_wmma_f32_16x16x32_bf16(
            false, a1.v, false, b.v, (short)0, acc2[1][t], false, false);
      }
    }
    __syncthreads();   // protect H/W2T before next chunk restages
  }

  // ---- sigmoid epilogue -> probs ----
#pragma unroll
  for (int s = 0; s < 2; ++s) {
#pragma unroll
    for (int t = 0; t < 4; ++t) {
      const int col = cw * 64 + t * 16 + colL;
      if (col < TOT) {
        const float bias = b2[col];
#pragma unroll
        for (int i = 0; i < 8; ++i) {
          int rg = row0 + 32 * rw + 16 * s + i + ((lane >> 4) & 1) * 8;
          float x = acc2[s][t][i] + bias;
          probs[(size_t)rg * TOT + col] = 1.0f / (1.0f + expf(-x));
        }
      }
    }
  }
}

// ---------------- valid_sum = sum(probs[0..63]) + probs[64] ----------------
__global__ void valid_kernel(const float* __restrict__ probs,
                             float* __restrict__ valid) {
  int row = blockIdx.x * 256 + threadIdx.x;
  if (row < ROWS) {
    const float* pr = probs + (size_t)row * TOT;
    float s = 0.f;
#pragma unroll
    for (int j = 0; j < 65; ++j) s += pr[j];
    valid[row] = s;
  }
}

// ---------------- tree DP: one block per (label, batch) ----------------
__global__ __launch_bounds__(128) void dp_kernel(const float* __restrict__ probs,
                                                 const float* __restrict__ valid,
                                                 float* __restrict__ out) {
  __shared__ float p[2][SEQL], q[2][SEQL];
  const int l = blockIdx.x;                // 0..63
  const int b = blockIdx.y;                // 0..127
  const int tid = threadIdx.x;
  const size_t base = (size_t)b * NODES;

  for (int i = tid; i < SEQL; i += 128) {
    float pt = probs[(base + i) * TOT + l];
    float vs = valid[base + i];
    p[0][i] = pt;
    q[0][i] = vs - pt;
  }
  __syncthreads();

  int off = SEQL, n = SEQL / 2, cur = 0;
  while (n >= 1) {
    for (int j = tid; j < n; j += 128) {
      size_t node = base + off + j;
      float ptn = probs[node * TOT + l];
      float nt  = probs[node * TOT + 65];
      float tv  = valid[node] - ptn;
      float lh = p[cur][2 * j],     rh = p[cur][2 * j + 1];
      float lo = q[cur][2 * j],     ro = q[cur][2 * j + 1];
      p[cur ^ 1][j] = ptn + nt * (lh * ro + lo * rh);
      q[cur ^ 1][j] = tv + nt * (lo * ro);
    }
    __syncthreads();
    off += n;
    n >>= 1;
    cur ^= 1;
  }
  if (tid == 0) out[(size_t)b * 64 + l] = p[cur][0];
}

// ---------------- host-side launch ----------------
extern "C" void kernel_launch(void* const* d_in, const int* in_sizes, int n_in,
                              void* d_out, int out_size, void* d_ws, size_t ws_size,
                              hipStream_t stream) {
  const float* emb = (const float*)d_in[0];
  const float* W1  = (const float*)d_in[1];
  const float* b1  = (const float*)d_in[2];
  const float* W2  = (const float*)d_in[3];
  const float* b2  = (const float*)d_in[4];
  float* out = (float*)d_out;

  char* ws = (char*)d_ws;
  float* probs          = (float*)(ws);                       // 34,569,216 B
  float* valid          = (float*)(ws + 34569216);            //    523,776 B
  unsigned short* Ebf   = (unsigned short*)(ws + 35092992);   // 201,129,984 B
  unsigned short* W1T   = (unsigned short*)(ws + 236222976);  //   4,718,592 B
  unsigned short* W2T   = (unsigned short*)(ws + 240941568);  //     786,432 B

  cvt_emb_kernel<<<8192, 256, 0, stream>>>(emb, Ebf, (long long)ROWS * EMB);
  cvt_w1t_kernel<<<(EMB * HID) / 256, 256, 0, stream>>>(W1, W1T);
  cvt_w2t_kernel<<<(128 * HID) / 256, 256, 0, stream>>>(W2, W2T);

  fused_mlp_kernel<<<ROWS / MT, 256, LDS_BYTES, stream>>>(Ebf, W1T, W2T, b1, b2, probs);

  valid_kernel<<<(ROWS + 255) / 256, 256, 0, stream>>>(probs, valid);
  dp_kernel<<<dim3(64, NBATCH), 128, 0, stream>>>(probs, valid, out);
}
